// HSTUBlock_50500225466720
// MI455X (gfx1250) — matrix operand order repaired and checked by reading, not compiled, not run on hardware
//
#include <hip/hip_runtime.h>

// HSTU block for gfx1250 (MI455X): wave32, WMMA bf16 + TDM (tensor_load_to_lds) path.
// Shapes: B=8, L=1024, D=512, H=8, DH=DL=64, NB=128.

#define B_ 8
#define L_ 1024
#define D_ 512
#define H_ 8
#define NB_ 128

typedef __attribute__((ext_vector_type(16))) __bf16 v16bf;
typedef __attribute__((ext_vector_type(8)))  float  v8f;

union Frag16 { unsigned u[8]; v16bf v; };

#if defined(__has_builtin)
#if __has_builtin(__builtin_amdgcn_tensor_load_to_lds) && __has_builtin(__builtin_amdgcn_s_wait_tensorcnt)
#define HAVE_TDM 1
#endif
#endif
#ifndef HAVE_TDM
#define HAVE_TDM 0
#endif

__device__ __forceinline__ unsigned short f2bf(float f) {
    unsigned u = __float_as_uint(f);
    u += 0x7FFFu + ((u >> 16) & 1u);      // round-to-nearest-even
    return (unsigned short)(u >> 16);
}
__device__ __forceinline__ float silu_f(float x) {
    // x * rcp(1+e^-x): single v_rcp_f32 instead of full IEEE divide chain
    return x * __builtin_amdgcn_rcpf(1.0f + __expf(-x));
}
__device__ __forceinline__ int imin_i(int a, int b) { return a < b ? a : b; }
__device__ __forceinline__ v8f v8zero() { v8f z = {0.f,0.f,0.f,0.f,0.f,0.f,0.f,0.f}; return z; }

__device__ __forceinline__ v8f wmma_bf16(Frag16 a, Frag16 b, v8f c) {
    return __builtin_amdgcn_wmma_f32_16x16x32_bf16(false, a.v, false, b.v, (short)0, c, false, false);
}

// A fragment 16x32 bf16 from row-major src[row][k] (ld in halfs).
__device__ __forceinline__ Frag16 load_a(const unsigned short* src, int row0, int ld, int k0, int lane) {
    Frag16 f;
    int m = lane & 15, hi8 = (lane >> 4) * 16;
    const unsigned short* p = src + (size_t)(row0 + m) * ld + k0;
#pragma unroll
    for (int r = 0; r < 8; ++r) {
        int k = 2 * r + hi8;
        f.u[r] = *(const unsigned*)(p + k);          // K-pairs contiguous -> b32 load
    }
    return f;
}

// B fragment 32x16 where B[k][n] = src[n][k] (src row-major [n][k], ld halfs).
__device__ __forceinline__ Frag16 load_b_t(const unsigned short* src, int n0, int ld, int k0, int lane) {
    Frag16 f;
    const unsigned short* p = src + (size_t)(n0 + (lane & 15)) * ld + k0 + (lane >> 4) * 16;
#pragma unroll
    for (int r = 0; r < 8; ++r) f.u[r] = *(const unsigned*)(p + 2 * r);  // contiguous b32
    return f;
}

#if HAVE_TDM
typedef unsigned int tdm_v4u __attribute__((ext_vector_type(4)));
typedef int          tdm_v8i __attribute__((ext_vector_type(8)));
typedef int          tdm_v4i __attribute__((ext_vector_type(4)));

__device__ __forceinline__ unsigned lds_offset(const void* p) {
    // AS3 pointers are byte offsets from the workgroup LDS base
    return (unsigned)(unsigned long long)(__attribute__((address_space(3))) const void*)p;
}

// 2-D TDM load: tile_d1 rows of tile_d0 bf16 elements, row stride stride0 (elements).
// Tensor dims == tile dims (tile start address given directly -> no OOB clipping needed).
__device__ __forceinline__ void tdm_load_2d(const void* gaddr, unsigned lds_off,
                                            unsigned tile_d0, unsigned tile_d1, unsigned stride0) {
    unsigned long long ga = (unsigned long long)gaddr;
    tdm_v4u g0;
    g0[0] = 1u;                                                   // count=1 (valid D#)
    g0[1] = lds_off;                                              // LDS byte address
    g0[2] = (unsigned)ga;                                         // global_addr[31:0]
    g0[3] = (unsigned)((ga >> 32) & 0x01FFFFFFu) | (2u << 30);    // global_addr[56:32] | type=2
    tdm_v8i g1;
    g1[0] = (int)(1u << 16);                                      // data_size=1 (2 bytes/elem)
    g1[1] = (int)((tile_d0 & 0xFFFFu) << 16);                     // tensor_dim0[15:0]
    g1[2] = (int)(((tile_d0 >> 16) & 0xFFFFu) | ((tile_d1 & 0xFFFFu) << 16)); // dim0 hi | dim1 lo
    g1[3] = (int)(((tile_d1 >> 16) & 0xFFFFu) | ((tile_d0 & 0xFFFFu) << 16)); // dim1 hi | tile_dim0
    g1[4] = (int)(tile_d1 & 0xFFFFu);                             // tile_dim1 (tile_dim2=0)
    g1[5] = (int)stride0;                                         // tensor_dim0_stride[31:0]
    g1[6] = 0; g1[7] = 0;
    tdm_v4i gz = {0, 0, 0, 0};
#if defined(__clang_major__) && __clang_major__ >= 23
    tdm_v8i gz8 = {0, 0, 0, 0, 0, 0, 0, 0};
    __builtin_amdgcn_tensor_load_to_lds(g0, g1, gz, gz, gz8, 0);
#else
    __builtin_amdgcn_tensor_load_to_lds(g0, g1, gz, gz, 0);
#endif
}
#endif // HAVE_TDM

// ---------------- LN1: x -> bf16 normalized ----------------
__global__ void ln1_kernel(const float* __restrict__ x, const float* __restrict__ g,
                           const float* __restrict__ bt, unsigned short* __restrict__ xn) {
    int row = blockIdx.x, tid = threadIdx.x;
    const float* xr = x + (size_t)row * D_;
    float a0 = xr[tid], a1 = xr[tid + 256];
    __shared__ float red[256];
    red[tid] = a0 + a1; __syncthreads();
    for (int s = 128; s > 0; s >>= 1) { if (tid < s) red[tid] += red[tid + s]; __syncthreads(); }
    float mean = red[0] * (1.0f / D_); __syncthreads();
    float d0 = a0 - mean, d1 = a1 - mean;
    red[tid] = d0 * d0 + d1 * d1; __syncthreads();
    for (int s = 128; s > 0; s >>= 1) { if (tid < s) red[tid] += red[tid + s]; __syncthreads(); }
    float rs = rsqrtf(red[0] * (1.0f / D_) + 1e-6f);
    unsigned short* o = xn + (size_t)row * D_;
    o[tid]       = f2bf(d0 * rs * g[tid]       + bt[tid]);
    o[tid + 256] = f2bf(d1 * rs * g[tid + 256] + bt[tid + 256]);
}

// ------------- cast + transpose weights to bf16 [n][k] -------------
__global__ void castw_kernel(const float* __restrict__ w1, const float* __restrict__ w2,
                             unsigned short* __restrict__ w1t, unsigned short* __restrict__ w2t) {
    int i = blockIdx.x * blockDim.x + threadIdx.x;       // 262144 threads
    for (int e = i; e < 2048 * 512; e += 262144) {
        int n = e / 512, k = e % 512;
        w1t[e] = f2bf(w1[(size_t)k * 2048 + n]);
    }
    {   // exactly 512*512 = 262144 elements
        int n = i / 512, k = i % 512;
        w2t[i] = f2bf(w2[(size_t)k * 512 + n]);
    }
}

// ------------- GEMM1: silu(xn @ W1) -> scatter u/v/q/k -------------
// grid (128, 8), block 256 (8 waves): wave tile 16x128, block tile 64x256
__global__ void gemm1_kernel(const unsigned short* __restrict__ xn, const unsigned short* __restrict__ w1t,
                             float* __restrict__ ubuf, unsigned short* __restrict__ qb,
                             unsigned short* __restrict__ kb, unsigned short* __restrict__ vb) {
    int lane = threadIdx.x & 31, wave = threadIdx.x >> 5;
    int r0 = blockIdx.x * 64 + (wave >> 1) * 16;
    int c0 = blockIdx.y * 256 + (wave & 1) * 128;
    v8f acc[8];
#pragma unroll
    for (int i = 0; i < 8; ++i) acc[i] = v8zero();

#if HAVE_TDM
    __shared__ unsigned short bw0[256 * 32];
    __shared__ unsigned short bw1[256 * 32];
    const unsigned short* wbase = w1t + (size_t)blockIdx.y * 256 * D_;
    if (wave == 0) tdm_load_2d(wbase, lds_offset(bw0), 32, 256, D_);
    for (int kc = 0; kc < 16; ++kc) {
        // prefetch next chunk into the other buffer (last iter: harmless dup into dead buffer)
        if (wave == 0)
            tdm_load_2d(wbase + imin_i(kc + 1, 15) * 32,
                        lds_offset((kc & 1) ? bw0 : bw1), 32, 256, D_);
        __builtin_amdgcn_s_wait_tensorcnt(1);   // current chunk resident
        __syncthreads();
        const unsigned short* bw = (kc & 1) ? bw1 : bw0;
        __builtin_prefetch(xn + (size_t)(r0 + (lane & 15)) * D_ + kc * 32 + 64, 0, 3);
        Frag16 a = load_a(xn, r0, D_, kc * 32, lane);
#pragma unroll
        for (int nf = 0; nf < 8; ++nf)
            acc[nf] = wmma_bf16(a, load_b_t(bw, (wave & 1) * 128 + nf * 16, 32, 0, lane), acc[nf]);
        __syncthreads();
    }
#else
    for (int k0 = 0; k0 < D_; k0 += 32) {
        __builtin_prefetch(xn + (size_t)(r0 + (lane & 15)) * D_ + k0 + 64, 0, 3);
        Frag16 a = load_a(xn, r0, D_, k0, lane);
#pragma unroll
        for (int nf = 0; nf < 8; ++nf)
            acc[nf] = wmma_bf16(a, load_b_t(w1t, c0 + nf * 16, D_, k0, lane), acc[nf]);
    }
#endif

#pragma unroll
    for (int nf = 0; nf < 8; ++nf) {
        int col = c0 + nf * 16 + (lane & 15);
#pragma unroll
        for (int r = 0; r < 8; ++r) {
            int row = r0 + r + 8 * (lane >> 4);
            float sv = silu_f(acc[nf][r]);
            int b = row >> 10, l = row & 1023;
            if (col < 512) {
                ubuf[(size_t)row * 512 + col] = sv;
            } else if (col < 1024) {
                int c = col - 512, h = c >> 6, dl = c & 63;   // V transposed: [b,h,dl,l]
                vb[((size_t)(b * H_ + h) * 64 + dl) * L_ + l] = f2bf(sv);
            } else if (col < 1536) {
                int c = col - 1024, h = c >> 6, d = c & 63;
                qb[(((size_t)(b * H_ + h)) * L_ + l) * 64 + d] = f2bf(sv);
            } else {
                int c = col - 1536, h = c >> 6, d = c & 63;
                kb[(((size_t)(b * H_ + h)) * L_ + l) * 64 + d] = f2bf(sv);
            }
        }
    }
}

// ------------- attention: silu(QK^T + bias)/L masked, @ V -------------
// grid (16 m-tiles, 64 bh), block 128 (4 waves; wave owns 16 rows of the 64-row tile)
__global__ void attn_kernel(const unsigned short* __restrict__ qb, const unsigned short* __restrict__ kb,
                            const unsigned short* __restrict__ vb, const int* __restrict__ ts,
                            const float* __restrict__ posw, const float* __restrict__ tsw,
                            float* __restrict__ attn) {
    __shared__ unsigned short lds_k[2][64 * 64];
    __shared__ unsigned short lds_v[2][64 * 64];
    __shared__ unsigned short lds_s[64 * 64];
    int tid = threadIdx.x, lane = tid & 31, wave = tid >> 5;
    int bh = blockIdx.y, b = bh >> 3, h = bh & 7;
    int mt = blockIdx.x, m0 = mt * 64;
    const unsigned short* qbase  = qb + (size_t)bh * L_ * 64;   // [l][dh]
    const unsigned short* kbase  = kb + (size_t)bh * L_ * 64;   // [l][dh]
    const unsigned short* vtbase = vb + (size_t)bh * 64 * L_;   // [dl][l]

    Frag16 aq0 = load_a(qbase, m0 + wave * 16, 64, 0, lane);
    Frag16 aq1 = load_a(qbase, m0 + wave * 16, 64, 32, lane);

    // timestamp row values depend only on m: hoist out of the n loop
    int tmv[8];
#pragma unroll
    for (int r = 0; r < 8; ++r) {
        int m_glob = m0 + wave * 16 + r + 8 * (lane >> 4);
        tmv[r] = ts[b * L_ + imin_i(m_glob + 1, L_ - 1)];
    }

    v8f o[4];
#pragma unroll
    for (int i = 0; i < 4; ++i) o[i] = v8zero();

#if HAVE_TDM
    if (wave == 0) tdm_load_2d(kbase,  lds_offset(lds_k[0]), 64, 64, 64);
    if (wave == 1) tdm_load_2d(vtbase, lds_offset(lds_v[0]), 64, 64, L_);
#endif

    for (int nt = 0; nt <= mt; ++nt) {
#if HAVE_TDM
        int cb = nt & 1, pb = (nt + 1) & 1;
        int pn = imin_i(nt + 1, mt);   // last iter: harmless dup into dead buffer
        if (wave == 0) tdm_load_2d(kbase + (size_t)pn * 64 * 64, lds_offset(lds_k[pb]), 64, 64, 64);
        if (wave == 1) tdm_load_2d(vtbase + (size_t)pn * 64,     lds_offset(lds_v[pb]), 64, 64, L_);
        __builtin_amdgcn_s_wait_tensorcnt(1);   // tile nt resident (per-wave in-order)
        __syncthreads();
#else
        int cb = 0;
        {
            const unsigned* sk = (const unsigned*)(kbase + (size_t)nt * 64 * 64);
            unsigned* dk = (unsigned*)lds_k[0];
            unsigned* dv = (unsigned*)lds_v[0];
#pragma unroll
            for (int i = 0; i < 16; ++i) {
                int idx = tid + i * 128;
                dk[idx] = sk[idx];
                int j = idx >> 5, c = idx & 31;   // V^T rows stride L_
                dv[j * 32 + c] = *(const unsigned*)(vtbase + (size_t)j * L_ + nt * 64 + 2 * c);
            }
        }
        __syncthreads();
#endif
        const unsigned short* lk = lds_k[cb];
        const unsigned short* lv = lds_v[cb];

        // S = Q K^T, bias + silu + mask in f32, re-layout via LDS as bf16
#pragma unroll
        for (int nf = 0; nf < 4; ++nf) {
            v8f c = v8zero();
            c = wmma_bf16(aq0, load_b_t(lk, nf * 16, 64, 0, lane), c);
            c = wmma_bf16(aq1, load_b_t(lk, nf * 16, 64, 32, lane), c);
            int n_loc = nf * 16 + (lane & 15);
            int n_glob = nt * 64 + n_loc;
            int tn = ts[b * L_ + n_glob];
#pragma unroll
            for (int r = 0; r < 8; ++r) {
                int m_loc = wave * 16 + r + 8 * (lane >> 4);
                int m_glob = m0 + m_loc;
                float sv = c[r];
                float pos = posw[1023 + n_glob - m_glob];
                float mag = fmaxf(fabsf((float)(tmv[r] - tn)), 1.0f);
                int bucket = (int)(__logf(mag) * (1.0f / 0.301f));
                bucket = bucket > NB_ ? NB_ : bucket;
                sv += pos + tsw[bucket];
                sv = silu_f(sv) * (1.0f / L_);
                if (n_glob > m_glob) sv = 0.0f;      // causal mask (triu k=1 zeroed)
                lds_s[m_loc * 64 + n_loc] = f2bf(sv);
            }
        }
        __syncthreads();
        // O += S @ V   (B[k][j] = V^T[j][k] -> contiguous dword LDS loads)
        Frag16 as0 = load_a(lds_s, wave * 16, 64, 0, lane);
        Frag16 as1 = load_a(lds_s, wave * 16, 64, 32, lane);
#pragma unroll
        for (int jf = 0; jf < 4; ++jf) {
            o[jf] = wmma_bf16(as0, load_b_t(lv, jf * 16, 64, 0, lane), o[jf]);
            o[jf] = wmma_bf16(as1, load_b_t(lv, jf * 16, 64, 32, lane), o[jf]);
        }
        __syncthreads();
    }
#pragma unroll
    for (int jf = 0; jf < 4; ++jf) {
        int col = h * 64 + jf * 16 + (lane & 15);
#pragma unroll
        for (int r = 0; r < 8; ++r) {
            int m_glob = m0 + wave * 16 + r + 8 * (lane >> 4);
            attn[((size_t)(b * L_ + m_glob)) * D_ + col] = o[jf][r];
        }
    }
}

// ------------- LN2 * u -> bf16 -------------
__global__ void ln2_kernel(const float* __restrict__ attn, const float* __restrict__ g,
                           const float* __restrict__ bt, const float* __restrict__ u,
                           unsigned short* __restrict__ an) {
    int row = blockIdx.x, tid = threadIdx.x;
    const float* xr = attn + (size_t)row * D_;
    float a0 = xr[tid], a1 = xr[tid + 256];
    __shared__ float red[256];
    red[tid] = a0 + a1; __syncthreads();
    for (int s = 128; s > 0; s >>= 1) { if (tid < s) red[tid] += red[tid + s]; __syncthreads(); }
    float mean = red[0] * (1.0f / D_); __syncthreads();
    float d0 = a0 - mean, d1 = a1 - mean;
    red[tid] = d0 * d0 + d1 * d1; __syncthreads();
    for (int s = 128; s > 0; s >>= 1) { if (tid < s) red[tid] += red[tid + s]; __syncthreads(); }
    float rs = rsqrtf(red[0] * (1.0f / D_) + 1e-6f);
    const float* ur = u + (size_t)row * D_;
    unsigned short* o = an + (size_t)row * D_;
    o[tid]       = f2bf((d0 * rs * g[tid]       + bt[tid])       * ur[tid]);
    o[tid + 256] = f2bf((d1 * rs * g[tid + 256] + bt[tid + 256]) * ur[tid + 256]);
}

// ------------- GEMM2: an @ out_w + out_b + x -> out -------------
// grid (128, 2), block 256: wave tile 16x128, block tile 64x256
__global__ void gemm2_kernel(const unsigned short* __restrict__ an, const unsigned short* __restrict__ w2t,
                             const float* __restrict__ ob, const float* __restrict__ x,
                             float* __restrict__ out) {
    int lane = threadIdx.x & 31, wave = threadIdx.x >> 5;
    int r0 = blockIdx.x * 64 + (wave >> 1) * 16;
    int c0 = blockIdx.y * 256 + (wave & 1) * 128;
    v8f acc[8];
#pragma unroll
    for (int i = 0; i < 8; ++i) acc[i] = v8zero();

#if HAVE_TDM
    __shared__ unsigned short bw0[256 * 32];
    __shared__ unsigned short bw1[256 * 32];
    const unsigned short* wbase = w2t + (size_t)blockIdx.y * 256 * D_;
    if (wave == 0) tdm_load_2d(wbase, lds_offset(bw0), 32, 256, D_);
    for (int kc = 0; kc < 16; ++kc) {
        if (wave == 0)
            tdm_load_2d(wbase + imin_i(kc + 1, 15) * 32,
                        lds_offset((kc & 1) ? bw0 : bw1), 32, 256, D_);
        __builtin_amdgcn_s_wait_tensorcnt(1);
        __syncthreads();
        const unsigned short* bw = (kc & 1) ? bw1 : bw0;
        __builtin_prefetch(an + (size_t)(r0 + (lane & 15)) * D_ + kc * 32 + 64, 0, 3);
        Frag16 a = load_a(an, r0, D_, kc * 32, lane);
#pragma unroll
        for (int nf = 0; nf < 8; ++nf)
            acc[nf] = wmma_bf16(a, load_b_t(bw, (wave & 1) * 128 + nf * 16, 32, 0, lane), acc[nf]);
        __syncthreads();
    }
#else
    for (int k0 = 0; k0 < D_; k0 += 32) {
        Frag16 a = load_a(an, r0, D_, k0, lane);
#pragma unroll
        for (int nf = 0; nf < 8; ++nf)
            acc[nf] = wmma_bf16(a, load_b_t(w2t, c0 + nf * 16, D_, k0, lane), acc[nf]);
    }
#endif

#pragma unroll
    for (int nf = 0; nf < 8; ++nf) {
        int col = c0 + nf * 16 + (lane & 15);
#pragma unroll
        for (int r = 0; r < 8; ++r) {
            int row = r0 + r + 8 * (lane >> 4);
            out[(size_t)row * D_ + col] = acc[nf][r] + ob[col] + x[(size_t)row * D_ + col];
        }
    }
}

extern "C" void kernel_launch(void* const* d_in, const int* in_sizes, int n_in,
                              void* d_out, int out_size, void* d_ws, size_t ws_size,
                              hipStream_t stream) {
    (void)in_sizes; (void)n_in; (void)out_size; (void)ws_size;
    const float* x      = (const float*)d_in[0];
    const int*   ts     = (const int*)d_in[1];
    // d_in[2] attn_mask: causal mask computed analytically
    const float* uvqk_w = (const float*)d_in[3];
    const float* out_w  = (const float*)d_in[4];
    const float* out_b  = (const float*)d_in[5];
    const float* ln1_g  = (const float*)d_in[6];
    const float* ln1_b  = (const float*)d_in[7];
    const float* ln2_g  = (const float*)d_in[8];
    const float* ln2_b  = (const float*)d_in[9];
    const float* tsw    = (const float*)d_in[10];
    const float* posw   = (const float*)d_in[11];
    float* out = (float*)d_out;

    // workspace layout (halfs then floats)
    unsigned short* xn  = (unsigned short*)d_ws;            // 8192*512
    unsigned short* w1t = xn  + (size_t)8192 * 512;         // 2048*512 (transposed [n][k])
    unsigned short* w2t = w1t + (size_t)2048 * 512;         // 512*512  (transposed [n][k])
    unsigned short* qb  = w2t + (size_t)512 * 512;          // 64*1024*64  [b,h,l,dh]
    unsigned short* kb  = qb  + (size_t)64 * 1024 * 64;     // [b,h,l,dh]
    unsigned short* vb  = kb  + (size_t)64 * 1024 * 64;     // [b,h,dl,l] (transposed)
    unsigned short* an  = vb  + (size_t)64 * 1024 * 64;     // 8192*512
    float* ubuf = (float*)(an + (size_t)8192 * 512);        // 8192*512 f32
    float* attn = ubuf + (size_t)8192 * 512;                // 8192*512 f32

    ln1_kernel  <<<8192, 256, 0, stream>>>(x, ln1_g, ln1_b, xn);
    castw_kernel<<<1024, 256, 0, stream>>>(uvqk_w, out_w, w1t, w2t);
    gemm1_kernel<<<dim3(128, 8), 256, 0, stream>>>(xn, w1t, ubuf, qb, kb, vb);
    attn_kernel <<<dim3(16, 64), 128, 0, stream>>>(qb, kb, vb, ts, posw, tsw, attn);
    ln2_kernel  <<<8192, 256, 0, stream>>>(attn, ln2_g, ln2_b, ubuf, an);
    gemm2_kernel<<<dim3(128, 2), 256, 0, stream>>>(an, w2t, out_b, x, out);
}
